// GCN_56109452754981
// MI455X (gfx1250) — compile-verified, matrix operand order checked by reading, not run on hardware
//
#include <hip/hip_runtime.h>
#include <hip/hip_bf16.h>

// ---------------------------------------------------------------------------
// GCN (deep+wide+attention fusion) for MI455X / gfx1250.
// All GEMMs use v_wmma_f32_16x16x32_bf16 (bf16 A/B, f32 accum, wave32).
// Message passing uses hardware global_atomic_add_f32 (unsafeAtomicAdd); the
// [N,128] operand matrices live in workspace and are L2-resident (51MB << 192MB L2).
// ---------------------------------------------------------------------------

typedef __attribute__((ext_vector_type(16))) __bf16 v16bf;
typedef __attribute__((ext_vector_type(4)))  __bf16 v4bf;
typedef __attribute__((ext_vector_type(8)))  float  v8f;

union FragBF {
    v16bf v;
    uint4 q[2];
    __bf16 e[16];
};

static __device__ inline v8f wmma_bf16(v16bf a, v16bf b, v8f c) {
    // 8 args: (neg_a, A, neg_b, B, c_mod, C, reuse_a, reuse_b)
    return __builtin_amdgcn_wmma_f32_16x16x32_bf16(false, a, false, b, (short)0, c, false, false);
}

// Hardware f32 atomic add (global_atomic_add_f32), not a CAS loop.
static __device__ inline void atomic_add_f32(float* p, float v) {
    unsafeAtomicAdd(p, v);
}

// A fragment (16x32, bf16) from a row-major bf16 matrix.
// lane&15 -> M row, lane>>4 -> K half. elems 0..7 = K kh*8+0..7, elems 8..15 = K 16+kh*8+0..7.
static __device__ inline v16bf load_fragA_bf16(const __bf16* base, int ld, int m, int kbase, int lane) {
    int kh = lane >> 4;
    const __bf16* p = base + (size_t)m * ld + kbase + kh * 8;
    FragBF f;
    f.q[0] = *(const uint4*)(p);
    f.q[1] = *(const uint4*)(p + 16);
    return f.v;
}

// A fragment from a row-major fp32 matrix, converted to bf16 on the fly.
static __device__ inline v16bf load_fragA_f32(const float* base, int ld, int m, int kbase, int lane) {
    int kh = lane >> 4;
    const float* p = base + (size_t)m * ld + kbase + kh * 8;
    float4 a = *(const float4*)(p);
    float4 b = *(const float4*)(p + 4);
    float4 c = *(const float4*)(p + 16);
    float4 d = *(const float4*)(p + 20);
    FragBF f;
    f.e[0] = (__bf16)a.x;  f.e[1] = (__bf16)a.y;  f.e[2] = (__bf16)a.z;  f.e[3] = (__bf16)a.w;
    f.e[4] = (__bf16)b.x;  f.e[5] = (__bf16)b.y;  f.e[6] = (__bf16)b.z;  f.e[7] = (__bf16)b.w;
    f.e[8] = (__bf16)c.x;  f.e[9] = (__bf16)c.y;  f.e[10] = (__bf16)c.z; f.e[11] = (__bf16)c.w;
    f.e[12] = (__bf16)d.x; f.e[13] = (__bf16)d.y; f.e[14] = (__bf16)d.z; f.e[15] = (__bf16)d.w;
    return f.v;
}

// B fragment from fragment-linear packed buffer: frag_idx blocks of 32 lanes x 16 bf16.
static __device__ inline v16bf load_fragB_packed(const __bf16* pk, int frag_idx, int lane) {
    const __bf16* p = pk + ((size_t)frag_idx * 32 + lane) * 16;
    FragBF f;
    f.q[0] = *(const uint4*)(p);
    f.q[1] = *(const uint4*)(p + 8);
    return f.v;
}

// ---------------------------------------------------------------------------
// Weight packing: combined [512 x 176] = W1(128) | Ww(40) | Wa(2) | pad(6)
// ---------------------------------------------------------------------------
__global__ void k_pack_combined(const float* __restrict__ W1, const float* __restrict__ Ww,
                                const float* __restrict__ Wa, __bf16* __restrict__ out) {
    const int NT = 11, KS = 16;
    int idx = blockIdx.x * blockDim.x + threadIdx.x;
    if (idx >= NT * KS * 32) return;
    int lane = idx & 31;
    int s = (idx >> 5) & (KS - 1);
    int t = idx >> 9;
    int n = t * 16 + (lane & 15);
    int kh = lane >> 4;
    __bf16* dst = out + (size_t)idx * 16;
#pragma unroll
    for (int i = 0; i < 16; i++) {
        int k = s * 32 + ((i < 8) ? 0 : 16) + kh * 8 + (i & 7);
        float v = 0.f;
        if (n < 128)      v = W1[(size_t)k * 128 + n];
        else if (n < 168) v = Ww[(size_t)k * 40 + (n - 128)];
        else if (n < 170) v = Wa[(size_t)k * 2 + (n - 168)];
        dst[i] = (__bf16)v;
    }
}

// Generic pack: W is [K x Ncols] row-major; output Ntiles col-tiles, K/32 k-steps.
__global__ void k_pack_w(const float* __restrict__ W, int K, int Ncols, int Ntiles,
                         __bf16* __restrict__ out) {
    int KS = K / 32;
    int total = Ntiles * KS * 32;
    int idx = blockIdx.x * blockDim.x + threadIdx.x;
    if (idx >= total) return;
    int lane = idx & 31;
    int s = (idx >> 5) % KS;
    int t = idx / (32 * KS);
    int n = t * 16 + (lane & 15);
    int kh = lane >> 4;
    __bf16* dst = out + (size_t)idx * 16;
#pragma unroll
    for (int i = 0; i < 16; i++) {
        int k = s * 32 + ((i < 8) ? 0 : 16) + kh * 8 + (i & 7);
        float v = (n < Ncols) ? W[(size_t)k * Ncols + n] : 0.f;
        dst[i] = (__bf16)v;
    }
}

// ---------------------------------------------------------------------------
// Degree / normalization
// ---------------------------------------------------------------------------
__global__ void k_deg_init(float* __restrict__ deg, int n) {
    int i = blockIdx.x * blockDim.x + threadIdx.x;
    if (i < n) deg[i] = 1.0f;  // self-loop weight
}

__global__ void k_deg_scatter(const long long* __restrict__ ei, const float* __restrict__ ew,
                              float* __restrict__ deg, int E) {
    int e = blockIdx.x * blockDim.x + threadIdx.x;
    if (e >= E) return;
    long long c = ei[(size_t)E + e];  // col = target
    atomic_add_f32(&deg[c], ew[e]);
}

__global__ void k_dinv(float* __restrict__ deg, int n) {
    int i = blockIdx.x * blockDim.x + threadIdx.x;
    if (i < n) {
        float d = deg[i];
        deg[i] = (d > 0.f) ? rsqrtf(d) : 0.f;  // in place: deg -> dinv
    }
}

// ---------------------------------------------------------------------------
// Fused input GEMM: h = relu(x@W1+b1), wide = x@Ww+bw, attn_logits = x@Wa+ba
// 11 waves per block, one 16-col tile each, M-tile per block. K = 512.
// ---------------------------------------------------------------------------
__global__ __launch_bounds__(352) void k_gemm1(const float* __restrict__ x,
                                               const __bf16* __restrict__ pkW,
                                               const float* __restrict__ b1,
                                               const float* __restrict__ bw,
                                               const float* __restrict__ ba,
                                               __bf16* __restrict__ h,
                                               float* __restrict__ wide,
                                               float* __restrict__ attn, int n_nodes) {
    const int KS = 16;
    int t = threadIdx.x >> 5;   // wave id = column tile (0..10)
    int lane = threadIdx.x & 31;
    int mt = blockIdx.x;
    int mload = mt * 16 + (lane & 15);
    if (mload > n_nodes - 1) mload = n_nodes - 1;
    v8f acc = {0.f, 0.f, 0.f, 0.f, 0.f, 0.f, 0.f, 0.f};
#pragma unroll 4
    for (int s = 0; s < KS; s++) {
        __builtin_prefetch(x + (size_t)mload * 512 + (s + 1) * 32, 0, 3);
        v16bf a = load_fragA_f32(x, 512, mload, s * 32, lane);
        v16bf b = load_fragB_packed(pkW, t * KS + s, lane);
        acc = wmma_bf16(a, b, acc);
    }
    int n = t * 16 + (lane & 15);
    int rbase = 8 * (lane >> 4);
    float bias = 0.f;
    if (n < 128) bias = b1[n];
    else if (n < 168) bias = bw[n - 128];
    else if (n < 170) bias = ba[n - 168];
#pragma unroll
    for (int i = 0; i < 8; i++) {
        int row = mt * 16 + rbase + i;
        if (row >= n_nodes) continue;
        float v = acc[i] + bias;
        if (n < 128) {
            h[(size_t)row * 128 + n] = (__bf16)fmaxf(v, 0.f);
        } else if (n < 168) {
            wide[(size_t)row * 40 + (n - 128)] = v;
        } else if (n < 170) {
            attn[(size_t)row * 2 + (n - 168)] = v;
        }
    }
}

// ---------------------------------------------------------------------------
// Conv GEMM + fused self-loop: xw = h @ Wc; agg = dinv^2 * xw  (8 waves/block)
// ---------------------------------------------------------------------------
__global__ __launch_bounds__(256) void k_gemm_conv(const __bf16* __restrict__ h,
                                                   const __bf16* __restrict__ pkW,
                                                   const float* __restrict__ dinv,
                                                   float* __restrict__ xw,
                                                   float* __restrict__ agg, int n_nodes) {
    int t = threadIdx.x >> 5;
    int lane = threadIdx.x & 31;
    int mt = blockIdx.x;
    int mload = mt * 16 + (lane & 15);
    if (mload > n_nodes - 1) mload = n_nodes - 1;
    v8f acc = {0.f, 0.f, 0.f, 0.f, 0.f, 0.f, 0.f, 0.f};
#pragma unroll
    for (int s = 0; s < 4; s++) {
        v16bf a = load_fragA_bf16(h, 128, mload, s * 32, lane);
        v16bf b = load_fragB_packed(pkW, t * 4 + s, lane);
        acc = wmma_bf16(a, b, acc);
    }
    int n = t * 16 + (lane & 15);
    int rbase = 8 * (lane >> 4);
#pragma unroll
    for (int i = 0; i < 8; i++) {
        int row = mt * 16 + rbase + i;
        if (row < n_nodes) {
            float v = acc[i];
            float di = dinv[row];
            xw[(size_t)row * 128 + n] = v;
            agg[(size_t)row * 128 + n] = di * di * v;  // self-loop term
        }
    }
}

// One wave per edge: agg[col] += (dinv[row]*w*dinv[col]) * xw[row]; 4 floats/lane.
__global__ __launch_bounds__(256) void k_edge_scatter(const long long* __restrict__ ei,
                                                      const float* __restrict__ ew,
                                                      const float* __restrict__ dinv,
                                                      const float* __restrict__ xw,
                                                      float* __restrict__ agg, int E) {
    int e = blockIdx.x * 8 + (threadIdx.x >> 5);
    if (e >= E) return;
    int lane = threadIdx.x & 31;
    long long r = ei[e];
    long long c = ei[(size_t)E + e];
    float coef = dinv[r] * ew[e] * dinv[c];
    const float4 v = *(const float4*)(xw + (size_t)r * 128 + lane * 4);
    float* dst = agg + (size_t)c * 128 + lane * 4;
    atomic_add_f32(dst + 0, coef * v.x);
    atomic_add_f32(dst + 1, coef * v.y);
    atomic_add_f32(dst + 2, coef * v.z);
    atomic_add_f32(dst + 3, coef * v.w);
}

// h[i,f] = bf16(relu(agg[i,f] + bc[f]))  -- vectorized 4 elems/thread
__global__ void k_postagg(const float* __restrict__ agg, const float* __restrict__ bc,
                          __bf16* __restrict__ h, int n_nodes) {
    int idx = blockIdx.x * blockDim.x + threadIdx.x;  // one per 4 features
    if (idx >= n_nodes * 32) return;
    int f4 = idx & 31;
    const float4 a = *(const float4*)(agg + (size_t)idx * 4);
    const float4 b = *(const float4*)(bc + f4 * 4);
    v4bf o;
    o[0] = (__bf16)fmaxf(a.x + b.x, 0.f);
    o[1] = (__bf16)fmaxf(a.y + b.y, 0.f);
    o[2] = (__bf16)fmaxf(a.z + b.z, 0.f);
    o[3] = (__bf16)fmaxf(a.w + b.w, 0.f);
    *(v4bf*)(h + (size_t)idx * 4) = o;
}

// ---------------------------------------------------------------------------
// Final: deep = h@W2+b2 (40 cols, padded 48); out = deep*a0 + wide*a1,
// (a0,a1) = softmax(attn_logits). 3 waves/block.
// ---------------------------------------------------------------------------
__global__ __launch_bounds__(96) void k_final(const __bf16* __restrict__ h,
                                              const __bf16* __restrict__ pkW2,
                                              const float* __restrict__ b2,
                                              const float* __restrict__ wide,
                                              const float* __restrict__ attnlog,
                                              float* __restrict__ out, int n_nodes) {
    int t = threadIdx.x >> 5;   // 0..2
    int lane = threadIdx.x & 31;
    int mt = blockIdx.x;
    int mload = mt * 16 + (lane & 15);
    if (mload > n_nodes - 1) mload = n_nodes - 1;
    v8f acc = {0.f, 0.f, 0.f, 0.f, 0.f, 0.f, 0.f, 0.f};
#pragma unroll
    for (int s = 0; s < 4; s++) {
        v16bf a = load_fragA_bf16(h, 128, mload, s * 32, lane);
        v16bf b = load_fragB_packed(pkW2, t * 4 + s, lane);
        acc = wmma_bf16(a, b, acc);
    }
    int n = t * 16 + (lane & 15);
    int rbase = 8 * (lane >> 4);
#pragma unroll
    for (int i = 0; i < 8; i++) {
        int row = mt * 16 + rbase + i;
        if (row >= n_nodes || n >= 40) continue;
        float d = acc[i] + b2[n];
        float l0 = attnlog[(size_t)row * 2 + 0];
        float l1 = attnlog[(size_t)row * 2 + 1];
        float mx = fmaxf(l0, l1);
        float e0 = __expf(l0 - mx);
        float e1 = __expf(l1 - mx);
        float inv = 1.f / (e0 + e1);
        out[(size_t)row * 40 + n] = d * (e0 * inv) + wide[(size_t)row * 40 + n] * (e1 * inv);
    }
}

// ---------------------------------------------------------------------------
extern "C" void kernel_launch(void* const* d_in, const int* in_sizes, int n_in,
                              void* d_out, int out_size, void* d_ws, size_t ws_size,
                              hipStream_t stream) {
    (void)n_in; (void)out_size; (void)ws_size;
    const float*     x   = (const float*)d_in[0];
    const long long* ei  = (const long long*)d_in[1];
    const float*     ew  = (const float*)d_in[2];
    const float*     W1  = (const float*)d_in[3];
    const float*     b1  = (const float*)d_in[4];
    const float*     Wc1 = (const float*)d_in[5];
    const float*     bc1 = (const float*)d_in[6];
    const float*     Wc2 = (const float*)d_in[7];
    const float*     bc2 = (const float*)d_in[8];
    const float*     W2  = (const float*)d_in[9];
    const float*     b2  = (const float*)d_in[10];
    const float*     Ww  = (const float*)d_in[11];
    const float*     bw  = (const float*)d_in[12];
    const float*     Wa  = (const float*)d_in[13];
    const float*     ba  = (const float*)d_in[14];

    const int N = in_sizes[0] / 512;
    const int E = in_sizes[2];
    float* out = (float*)d_out;

    // ---- workspace layout (256B aligned slices) ----
    char* base = (char*)d_ws;
    size_t off = 0;
    auto take = [&](size_t bytes) -> char* {
        char* p = base + off;
        off = (off + bytes + 255) & ~(size_t)255;
        return p;
    };
    __bf16* hbuf   = (__bf16*)take((size_t)N * 128 * 2);
    float*  xw     = (float*)take((size_t)N * 128 * 4);
    float*  agg    = (float*)take((size_t)N * 128 * 4);
    float*  wide   = (float*)take((size_t)N * 40 * 4);
    float*  attn   = (float*)take((size_t)N * 2 * 4);
    float*  dinv   = (float*)take((size_t)N * 4);          // deg then dinv in place
    __bf16* pkComb = (__bf16*)take((size_t)11 * 16 * 32 * 16 * 2);
    __bf16* pkC1   = (__bf16*)take((size_t)8 * 4 * 32 * 16 * 2);
    __bf16* pkC2   = (__bf16*)take((size_t)8 * 4 * 32 * 16 * 2);
    __bf16* pkW2   = (__bf16*)take((size_t)3 * 4 * 32 * 16 * 2);

    const int Mtiles = (N + 15) / 16;

    // ---- weight packing ----
    k_pack_combined<<<(11 * 16 * 32 + 255) / 256, 256, 0, stream>>>(W1, Ww, Wa, pkComb);
    k_pack_w<<<(8 * 4 * 32 + 255) / 256, 256, 0, stream>>>(Wc1, 128, 128, 8, pkC1);
    k_pack_w<<<(8 * 4 * 32 + 255) / 256, 256, 0, stream>>>(Wc2, 128, 128, 8, pkC2);
    k_pack_w<<<(3 * 4 * 32 + 255) / 256, 256, 0, stream>>>(W2, 128, 40, 3, pkW2);

    // ---- degree / dinv ----
    k_deg_init<<<(N + 255) / 256, 256, 0, stream>>>(dinv, N);
    k_deg_scatter<<<(E + 255) / 256, 256, 0, stream>>>(ei, ew, dinv, E);
    k_dinv<<<(N + 255) / 256, 256, 0, stream>>>(dinv, N);

    // ---- fused input GEMM (lin1 + wide + attn logits) ----
    k_gemm1<<<Mtiles, 352, 0, stream>>>(x, pkComb, b1, bw, ba, hbuf, wide, attn, N);

    // ---- conv layer 1 ----
    k_gemm_conv<<<Mtiles, 256, 0, stream>>>(hbuf, pkC1, dinv, xw, agg, N);
    k_edge_scatter<<<(E + 7) / 8, 256, 0, stream>>>(ei, ew, dinv, xw, agg, E);
    k_postagg<<<((size_t)N * 32 + 255) / 256, 256, 0, stream>>>(agg, bc1, hbuf, N);

    // ---- conv layer 2 ----
    k_gemm_conv<<<Mtiles, 256, 0, stream>>>(hbuf, pkC2, dinv, xw, agg, N);
    k_edge_scatter<<<(E + 7) / 8, 256, 0, stream>>>(ei, ew, dinv, xw, agg, E);
    k_postagg<<<((size_t)N * 32 + 255) / 256, 256, 0, stream>>>(agg, bc2, hbuf, N);

    // ---- final GEMM + attention fusion ----
    k_final<<<Mtiles, 96, 0, stream>>>(hbuf, pkW2, b2, wide, attn, out, N);
}
